// AxialAttention_16441134809831
// MI455X (gfx1250) — matrix operand order repaired and checked by reading, not compiled, not run on hardware
//
#include <hip/hip_runtime.h>
#include <hip/hip_bf16.h>

// ---------------------------------------------------------------------------
// Axial attention for MI455X (gfx1250, wave32, WMMA).
//
//   1) convert weights fp32 -> bf16
//   2) qp/kp/vp = X @ W^T   (v_wmma_f32_16x16x32_bf16, A converted on the fly)
//   3) per (b, head, axial-row): 65x65 causal attention; Q/K staged into LDS
//      via async LDS-DMA (global_load_async_to_lds_b128) when available,
//      softmax in fp32 registers, both attention GEMMs via WMMA
//   4) out = ctx @ wo^T + bo
//
// All global stores go through per-wave LDS staging so they issue as
// global_store_b128 instead of scattered b16/b32 stores.
// ---------------------------------------------------------------------------

typedef __bf16 bf16_t;
typedef __attribute__((ext_vector_type(16))) __bf16 v16bf;
typedef __attribute__((ext_vector_type(8)))  __bf16 v8bf;
typedef __attribute__((ext_vector_type(8)))  float  v8f;
typedef __attribute__((ext_vector_type(4)))  float  v4f;

#if __has_builtin(__builtin_amdgcn_global_load_async_to_lds_b128) && \
    __has_builtin(__builtin_amdgcn_s_wait_asynccnt)
#define USE_ASYNC_LDS 1
#else
#define USE_ASYNC_LDS 0
#endif

namespace {
constexpr int kB      = 8;
constexpr int kSeq    = 4097;           // 1 + 64*64
constexpr int kDim    = 512;
constexpr int kHeads  = 8;
constexpr int kDHead  = 64;
constexpr int kRows   = 64;             // h = w = 64
constexpr int kM      = kB * kSeq;      // 32776 rows for the projections
constexpr float kScale = 0.125f;        // 64^-0.5
constexpr float kNegBig = -3.0e38f;
}

#if USE_ASYNC_LDS
// builtin signature (from hipcc diagnostic): arg0 = addrspace(1) v4i*,
// arg1 = LDS-side v4i*, arg2/arg3 = immediate offset / cache policy.
typedef int v4i_vs __attribute__((vector_size(16)));
typedef __attribute__((address_space(1))) v4i_vs* gv4i_p;
typedef __attribute__((address_space(3))) v4i_vs* lv4i_p;

__device__ __forceinline__ void async_ld16(const bf16_t* g, bf16_t* l) {
  __builtin_amdgcn_global_load_async_to_lds_b128((gv4i_p)g, (lv4i_p)l, 0, 0);
}
#endif

// ---------------- fp32 -> bf16 elementwise convert -------------------------
__global__ void cvt_f32_to_bf16_kernel(const float* __restrict__ src,
                                       bf16_t* __restrict__ dst, int n) {
  int i = blockIdx.x * blockDim.x + threadIdx.x;
  if (i < n) dst[i] = (bf16_t)src[i];
}

// ---------------- GEMM: Y(bf16)[M,512] = X(f32)[M,512] @ W(bf16)[512,512]^T
// 256 threads = 8 waves; wave owns a 16(M) x 64(N) strip, block 128 x 64.
__global__ __launch_bounds__(256)
void proj_gemm_f32a_kernel(const float* __restrict__ X,
                           const bf16_t* __restrict__ Wb,
                           bf16_t* __restrict__ Y, int M) {
  const int lane  = threadIdx.x & 31;
  const int wave  = threadIdx.x >> 5;
  const int half  = lane >> 4;
  const int l16   = lane & 15;
  const int mBase = blockIdx.x * 128 + wave * 16;
  const int nBase = blockIdx.y * 64;

  __shared__ alignas(16) bf16_t sC[8][16 * 64];   // per-wave C staging

  int arow = mBase + l16;
  if (arow >= M) arow = M - 1;                    // clamp (stores guarded)
  const float* aptr = X + (size_t)arow * kDim;

  v8f acc[4] = {v8f{}, v8f{}, v8f{}, v8f{}};

  for (int k0 = 0; k0 < kDim; k0 += 32) {
    __builtin_prefetch(aptr + k0 + 128, 0, 0);    // global_prefetch_b8
    v16bf afrag;                                  // A: two 8-elem K segments
    {
      const int s0 = k0 + half * 16;
      const int s1 = k0 + half * 16 + 8;
#pragma unroll
      for (int t = 0; t < 8; ++t) afrag[t]     = (bf16_t)aptr[s0 + t];
#pragma unroll
      for (int t = 0; t < 8; ++t) afrag[8 + t] = (bf16_t)aptr[s1 + t];
    }
#pragma unroll
    for (int nt = 0; nt < 4; ++nt) {              // B: lane = N col
      const bf16_t* bptr =
          Wb + (size_t)(nBase + nt * 16 + l16) * kDim + k0 + half * 16;
      v16bf bfrag;
#pragma unroll
      for (int t = 0; t < 16; ++t) bfrag[t] = bptr[t];
      acc[nt] = __builtin_amdgcn_wmma_f32_16x16x32_bf16(
          false, afrag, false, bfrag, (short)0, acc[nt], false, false);
    }
  }
  // C layout -> row-major LDS tile (same wave, LDS is in-order: no barrier)
#pragma unroll
  for (int nt = 0; nt < 4; ++nt)
#pragma unroll
    for (int g = 0; g < 8; ++g)
      sC[wave][(g + half * 8) * 64 + nt * 16 + l16] = (bf16_t)acc[nt][g];

  // vectorized store: 8 bf16 (16 B) per chunk, one bound check per chunk
#pragma unroll
  for (int e = lane * 8; e < 16 * 64; e += 32 * 8) {
    const int r = e >> 6, c = e & 63;
    const int row = mBase + r;
    if (row < M)
      *(v8bf*)(Y + (size_t)row * kDim + nBase + c) =
          *(const v8bf*)(&sC[wave][e]);
  }
}

// ---------------- GEMM: out(f32)[M,512] = ctx(bf16) @ wo(bf16)^T + bo ------
__global__ __launch_bounds__(256)
void out_gemm_kernel(const bf16_t* __restrict__ Xb,
                     const bf16_t* __restrict__ Wb,
                     const float* __restrict__ bias,
                     float* __restrict__ Y, int M) {
  const int lane  = threadIdx.x & 31;
  const int wave  = threadIdx.x >> 5;
  const int half  = lane >> 4;
  const int l16   = lane & 15;
  const int mBase = blockIdx.x * 128 + wave * 16;
  const int nBase = blockIdx.y * 64;

  __shared__ alignas(16) float sC[8][16 * 64];    // per-wave fp32 C staging

  int arow = mBase + l16;
  if (arow >= M) arow = M - 1;
  const bf16_t* aptr = Xb + (size_t)arow * kDim;

  v8f acc[4] = {v8f{}, v8f{}, v8f{}, v8f{}};

  for (int k0 = 0; k0 < kDim; k0 += 32) {
    __builtin_prefetch(aptr + k0 + 256, 0, 0);
    v16bf afrag;
    {
      const int s0 = k0 + half * 16;
      const int s1 = k0 + half * 16 + 8;
#pragma unroll
      for (int t = 0; t < 8; ++t) afrag[t]     = aptr[s0 + t];
#pragma unroll
      for (int t = 0; t < 8; ++t) afrag[8 + t] = aptr[s1 + t];
    }
#pragma unroll
    for (int nt = 0; nt < 4; ++nt) {
      const bf16_t* bptr =
          Wb + (size_t)(nBase + nt * 16 + l16) * kDim + k0 + half * 16;
      v16bf bfrag;
#pragma unroll
      for (int t = 0; t < 16; ++t) bfrag[t] = bptr[t];
      acc[nt] = __builtin_amdgcn_wmma_f32_16x16x32_bf16(
          false, afrag, false, bfrag, (short)0, acc[nt], false, false);
    }
  }
  // fold bias while restriping into LDS
#pragma unroll
  for (int nt = 0; nt < 4; ++nt) {
    const float bv = bias[nBase + nt * 16 + l16];
#pragma unroll
    for (int g = 0; g < 8; ++g)
      sC[wave][(g + half * 8) * 64 + nt * 16 + l16] = acc[nt][g] + bv;
  }
  // vectorized fp32 store: 4 floats (16 B) per chunk
#pragma unroll
  for (int e = lane * 4; e < 16 * 64; e += 32 * 4) {
    const int r = e >> 6, c = e & 63;
    const int row = mBase + r;
    if (row < M)
      *(v4f*)(Y + (size_t)row * kDim + nBase + c) =
          *(const v4f*)(&sC[wave][e]);
  }
}

// ---------------- axial attention: one wave per (b, head, row) -------------
// 65 tokens (bos + 64 columns), causal, dhead = 64.
// Padding: queries/keys to 80 (5x16 tiles), P/V token axis to 96 (3x32 steps).
__global__ __launch_bounds__(32)
void axial_attn_kernel(const bf16_t* __restrict__ qp,
                       const bf16_t* __restrict__ kp,
                       const bf16_t* __restrict__ vp,
                       bf16_t* __restrict__ ctx) {
  const int idx  = blockIdx.x;          // b*512 + head*64 + row
  const int row  = idx & 63;
  const int head = (idx >> 6) & 7;
  const int b    = idx >> 9;
  const int lane = threadIdx.x;
  const int half = lane >> 4;
  const int l16  = lane & 15;

  __shared__ alignas(16) bf16_t sQ[80][64];   // (query token, d)
  __shared__ alignas(16) bf16_t sK[80][64];   // (key token, d)
  __shared__ alignas(16) bf16_t sVt[64][96];  // (d, key token) transposed
  __shared__ alignas(16) bf16_t sP[16][96];   // probs tile, padded K
  __shared__ alignas(16) bf16_t sO[16][64];   // context tile staging

  // zero ONLY the padding regions (must not race the async DMA below)
  for (int e = lane; e < 15 * 64; e += 32) {  // sQ/sK rows 65..79
    (&sQ[65][0])[e] = (bf16_t)0.f;
    (&sK[65][0])[e] = (bf16_t)0.f;
  }
  for (int e = lane; e < 64 * 31; e += 32) {  // sVt tokens 65..95
    const int d = e / 31, t = 65 + (e % 31);
    sVt[d][t] = (bf16_t)0.f;
  }
  for (int e = lane; e < 16 * 16; e += 32)    // sP cols 80..95
    sP[e >> 4][80 + (e & 15)] = (bf16_t)0.f;

  const size_t base = (size_t)b * kSeq * kDim + (size_t)head * kDHead;

#if USE_ASYNC_LDS
  // Q/K: 65 tokens x 32 bf16 (d 0..31) = 65 x 4 chunks of 16B, async DMA to LDS
  for (int e = lane; e < 65 * 4; e += 32) {
    const int t = e >> 2, c = (e & 3) * 8;
    const size_t seq = (t == 0) ? 0 : (size_t)(1 + row * 64 + (t - 1));
    const size_t off = base + seq * kDim + c;
    async_ld16(qp + off, &sQ[t][c]);
    async_ld16(kp + off, &sK[t][c]);
  }
#endif
  // V: needs transpose, stage through VGPRs
  for (int e = lane; e < 65 * 64; e += 32) {
    const int t = e >> 6, d = e & 63;
    const size_t seq = (t == 0) ? 0 : (size_t)(1 + row * 64 + (t - 1));
    sVt[d][t] = vp[base + seq * kDim + d];
#if !USE_ASYNC_LDS
    sQ[t][d] = qp[base + seq * kDim + d];
    sK[t][d] = kp[base + seq * kDim + d];
#endif
  }
#if USE_ASYNC_LDS
  __builtin_amdgcn_s_wait_asynccnt(0);
#endif
  __syncthreads();

  for (int mt = 0; mt < 5; ++mt) {
    // ---- Q A-fragments for the two K(d) steps
    v16bf qa0, qa1;
    {
      const bf16_t* qrow = &sQ[mt * 16 + l16][0];
#pragma unroll
      for (int t = 0; t < 8; ++t) {
        qa0[t]     = qrow[half * 16 + t];
        qa0[8 + t] = qrow[half * 16 + 8 + t];
        qa1[t]     = qrow[32 + half * 16 + t];
        qa1[8 + t] = qrow[32 + half * 16 + 8 + t];
      }
    }
    // ---- dots = Q @ K^T  (5 key tiles)
    v8f dots[5];
#pragma unroll
    for (int nt = 0; nt < 5; ++nt) {
      const bf16_t* krow = &sK[nt * 16 + l16][0];
      v16bf kb0, kb1;
#pragma unroll
      for (int t = 0; t < 16; ++t) {
        kb0[t] = krow[half * 16 + t];
        kb1[t] = krow[32 + half * 16 + t];
      }
      v8f a{};
      a = __builtin_amdgcn_wmma_f32_16x16x32_bf16(false, qa0, false, kb0,
                                                  (short)0, a, false, false);
      a = __builtin_amdgcn_wmma_f32_16x16x32_bf16(false, qa1, false, kb1,
                                                  (short)0, a, false, false);
      dots[nt] = a;
    }
    // ---- causal mask + scale + softmax (each VGPR g / lane-half is one row)
    float p[5][8];
#pragma unroll
    for (int g = 0; g < 8; ++g) {
      const int i = mt * 16 + g + half * 8;
      float rm = kNegBig;
#pragma unroll
      for (int nt = 0; nt < 5; ++nt) {
        const int j = nt * 16 + l16;
        float v = dots[nt][g] * kScale;
        if (j > i || j >= 65) v = kNegBig;
        p[nt][g] = v;
        rm = fmaxf(rm, v);
      }
#pragma unroll
      for (int off = 8; off >= 1; off >>= 1)     // xor<16 stays in-half
        rm = fmaxf(rm, __shfl_xor(rm, off, 32));
      float sum = 0.f;
#pragma unroll
      for (int nt = 0; nt < 5; ++nt) {
        float e = __expf(p[nt][g] - rm);
        p[nt][g] = e;
        sum += e;
      }
#pragma unroll
      for (int off = 8; off >= 1; off >>= 1)
        sum += __shfl_xor(sum, off, 32);
      const float inv = 1.f / sum;
#pragma unroll
      for (int nt = 0; nt < 5; ++nt) p[nt][g] *= inv;
    }
    // ---- restripe P into A-operand layout via LDS (cols 80..95 stay zero)
#pragma unroll
    for (int g = 0; g < 8; ++g) {
      const int rloc = g + half * 8;
#pragma unroll
      for (int nt = 0; nt < 5; ++nt)
        sP[rloc][nt * 16 + l16] = (bf16_t)p[nt][g];
    }
    // single wave: same-wave LDS ops are in-order, no barrier needed

    // ---- out = P @ V  (K = 96 padded tokens, 3 steps; 4 output d-tiles)
    v8f o[4] = {v8f{}, v8f{}, v8f{}, v8f{}};
#pragma unroll
    for (int ks = 0; ks < 96; ks += 32) {
      v16bf pa;
      const bf16_t* prow = &sP[l16][0];
#pragma unroll
      for (int t = 0; t < 8; ++t) {
        pa[t]     = prow[ks + half * 16 + t];
        pa[8 + t] = prow[ks + half * 16 + 8 + t];
      }
#pragma unroll
      for (int ot = 0; ot < 4; ++ot) {
        const bf16_t* vrow = &sVt[ot * 16 + l16][0];
        v16bf vb;
#pragma unroll
        for (int t = 0; t < 16; ++t) vb[t] = vrow[ks + half * 16 + t];
        o[ot] = __builtin_amdgcn_wmma_f32_16x16x32_bf16(
            false, pa, false, vb, (short)0, o[ot], false, false);
      }
    }
    // ---- stage context tile row-major, then vectorized global stores
#pragma unroll
    for (int ot = 0; ot < 4; ++ot)
#pragma unroll
      for (int g = 0; g < 8; ++g)
        sO[g + half * 8][ot * 16 + l16] = (bf16_t)o[ot][g];

#pragma unroll
    for (int e = lane * 8; e < 16 * 64; e += 32 * 8) {
      const int r = e >> 6, c = e & 63;
      const int qt = mt * 16 + r;
      if (qt <= 64 && (qt >= 1 || row == 0)) {
        const size_t seq = (qt == 0) ? 0 : (size_t)(1 + row * 64 + (qt - 1));
        *(v8bf*)(ctx + base + seq * kDim + c) = *(const v8bf*)(&sO[r][c]);
      }
    }
  }
}

// ---------------------------------------------------------------------------
extern "C" void kernel_launch(void* const* d_in, const int* in_sizes, int n_in,
                              void* d_out, int out_size, void* d_ws,
                              size_t ws_size, hipStream_t stream) {
  const float* q  = (const float*)d_in[0];
  const float* k  = (const float*)d_in[1];
  const float* v  = (const float*)d_in[2];
  const float* wq = (const float*)d_in[3];
  const float* wk = (const float*)d_in[4];
  const float* wv = (const float*)d_in[5];
  const float* wo = (const float*)d_in[6];
  const float* bo = (const float*)d_in[7];
  float* out = (float*)d_out;

  char* ws = (char*)d_ws;
  const size_t wBytes   = (size_t)kDim * kDim * sizeof(bf16_t);   // 512 KB
  const size_t actBytes = (size_t)kM * kDim * sizeof(bf16_t);     // ~32 MB
  bf16_t* wqb = (bf16_t*)ws;  ws += wBytes;
  bf16_t* wkb = (bf16_t*)ws;  ws += wBytes;
  bf16_t* wvb = (bf16_t*)ws;  ws += wBytes;
  bf16_t* wob = (bf16_t*)ws;  ws += wBytes;
  bf16_t* qp  = (bf16_t*)ws;  ws += actBytes;
  bf16_t* kp  = (bf16_t*)ws;  ws += actBytes;
  bf16_t* vp  = (bf16_t*)ws;  ws += actBytes;
  bf16_t* ctx = (bf16_t*)ws;  ws += actBytes;

  const int wn = kDim * kDim;
  cvt_f32_to_bf16_kernel<<<(wn + 255) / 256, 256, 0, stream>>>(wq, wqb, wn);
  cvt_f32_to_bf16_kernel<<<(wn + 255) / 256, 256, 0, stream>>>(wk, wkb, wn);
  cvt_f32_to_bf16_kernel<<<(wn + 255) / 256, 256, 0, stream>>>(wv, wvb, wn);
  cvt_f32_to_bf16_kernel<<<(wn + 255) / 256, 256, 0, stream>>>(wo, wob, wn);

  dim3 gemmGrid((kM + 127) / 128, kDim / 64);  // 257 x 8
  proj_gemm_f32a_kernel<<<gemmGrid, 256, 0, stream>>>(q, wqb, qp, kM);
  proj_gemm_f32a_kernel<<<gemmGrid, 256, 0, stream>>>(k, wkb, kp, kM);
  proj_gemm_f32a_kernel<<<gemmGrid, 256, 0, stream>>>(v, wvb, vp, kM);

  axial_attn_kernel<<<kB * kHeads * kRows, 32, 0, stream>>>(qp, kp, vp, ctx);

  out_gemm_kernel<<<gemmGrid, 256, 0, stream>>>(ctx, wob, bo, out, kM);
}
